// SFC_module_17995912970298
// MI455X (gfx1250) — compile-verified, matrix operand order checked by reading, not run on hardware
//
#include <hip/hip_runtime.h>
#include <hip/hip_bf16.h>
#include <stdint.h>

typedef __bf16 bf16_t;
typedef __attribute__((ext_vector_type(16))) __bf16 v16bf;
typedef __attribute__((ext_vector_type(8)))  float  v8f;

#define Bn 8192
#define Dn 512
#define K4 2048          // concatenated K = 4*D
#define TILE 128
#define KSTEP 64
#define NITER (K4 / KSTEP)     // 32 stages
#define LDS_STRIDE 72          // 64 + 8 bf16 pad (144B rows, 16B-aligned segs)

// ---------------- kernel 1: build X/Y (bf16) + per-row stats ----------------
// X_i = [a, iv, mu, e],  Y_i = [iv, a, -2e, -2mu],  a = var+mu^2, e = mu*iv
// s_i = sum mu^2*iv ; P_i = prod var (exp of log-sum) ; rd_i = 1/(P_i+1e-8)
__global__ __launch_bounds__(256) void prep_kernel(
    const float* __restrict__ mu, const float* __restrict__ var,
    bf16_t* __restrict__ Xbf, bf16_t* __restrict__ Ybf,
    float* __restrict__ s_out, float* __restrict__ P_out,
    float* __restrict__ rd_out) {
  const int row = blockIdx.x;
  const int tid = threadIdx.x;
  const float* mrow = mu + (size_t)row * Dn;
  const float* vrow = var + (size_t)row * Dn;
  bf16_t* xr = Xbf + (size_t)row * K4;
  bf16_t* yr = Ybf + (size_t)row * K4;
  float s_acc = 0.f, lp_acc = 0.f;
  for (int d = tid; d < Dn; d += 256) {
    float m  = mrow[d];
    float v  = vrow[d];
    float iv = 1.0f / v;
    float m2 = m * m;
    float a  = v + m2;
    float e  = m * iv;
    xr[d]        = (bf16_t)a;
    xr[Dn + d]   = (bf16_t)iv;
    xr[2*Dn + d] = (bf16_t)m;
    xr[3*Dn + d] = (bf16_t)e;
    yr[d]        = (bf16_t)iv;
    yr[Dn + d]   = (bf16_t)a;
    yr[2*Dn + d] = (bf16_t)(-2.0f * e);
    yr[3*Dn + d] = (bf16_t)(-2.0f * m);
    s_acc  += m2 * iv;
    lp_acc += __logf(v);
  }
  __shared__ float red0[256];
  __shared__ float red1[256];
  red0[tid] = s_acc; red1[tid] = lp_acc;
  __syncthreads();
  for (int off = 128; off > 0; off >>= 1) {
    if (tid < off) { red0[tid] += red0[tid + off]; red1[tid] += red1[tid + off]; }
    __syncthreads();
  }
  if (tid == 0) {
    float P = __expf(red1[0]);
    s_out[row]  = red0[0];
    P_out[row]  = P;
    rd_out[row] = 1.0f / (P + 1e-8f);
  }
}

// ---------------- kernel 2: double-buffered bf16 WMMA GEMM + epilogue -------
union FragU { uint4 u[2]; v16bf v; };

__global__ __launch_bounds__(256) void gemm_kernel(
    const bf16_t* __restrict__ Xbf, const bf16_t* __restrict__ Ybf,
    const float* __restrict__ srow, const float* __restrict__ Prow,
    const float* __restrict__ rdrow, const int* __restrict__ labels,
    const float* __restrict__ alpha_p, const float* __restrict__ beta_p,
    float* __restrict__ out) {
  __shared__ __align__(16) bf16_t Atile[2][TILE * LDS_STRIDE];
  __shared__ __align__(16) bf16_t Btile[2][TILE * LDS_STRIDE];

  const int tid  = threadIdx.x;
  const int lane = tid & 31;
  const int wave = tid >> 5;
  const int half = lane >> 4;     // ISA 16-bit operand lane-half
  const int l16  = lane & 15;
  const int wmg  = wave & 3;      // wave M group: *32
  const int wng  = wave >> 2;     // wave N group: *64

  const int m0 = blockIdx.y * TILE;
  const int n0 = blockIdx.x * TILE;

  // per-thread global->LDS chunk assignment: 4 x 16B chunks per matrix/stage
  // chunk c in [0,1024): row = c>>3, seg (8 bf16) = c&7
  const bf16_t* gA[4];
  const bf16_t* gB[4];
  int lofs[4];
#pragma unroll
  for (int t = 0; t < 4; ++t) {
    int c  = tid + t * 256;
    int r  = c >> 3;
    int sg = c & 7;
    gA[t]   = Xbf + (size_t)(m0 + r) * K4 + sg * 8;
    gB[t]   = Ybf + (size_t)(n0 + r) * K4 + sg * 8;
    lofs[t] = r * LDS_STRIDE + sg * 8;
  }

  v8f acc[2][4];
#pragma unroll
  for (int i = 0; i < 2; ++i)
#pragma unroll
    for (int j = 0; j < 4; ++j)
      acc[i][j] = (v8f){0.f,0.f,0.f,0.f,0.f,0.f,0.f,0.f};

  // prologue: stage 0 -> buffer 0
  uint4 va[4], vb[4];
#pragma unroll
  for (int t = 0; t < 4; ++t) {
    va[t] = *(const uint4*)(gA[t]);
    vb[t] = *(const uint4*)(gB[t]);
  }
#pragma unroll
  for (int t = 0; t < 4; ++t) {
    *(uint4*)(&Atile[0][lofs[t]]) = va[t];
    *(uint4*)(&Btile[0][lofs[t]]) = vb[t];
  }
  __syncthreads();

  for (int it = 0; it < NITER; ++it) {
    const int cur = it & 1;
    const bool has_next = (it + 1 < NITER);

    // 1) kick off next stage's global loads (latency hides under WMMAs)
    if (has_next) {
      const int ko = (it + 1) * KSTEP;
#pragma unroll
      for (int t = 0; t < 4; ++t) {
        va[t] = *(const uint4*)(gA[t] + ko);
        vb[t] = *(const uint4*)(gB[t] + ko);
      }
    }

    // 2) consume current buffer: 2 K-substeps x (2 M x 4 N) WMMAs
#pragma unroll
    for (int s = 0; s < 2; ++s) {
      // frag elem e -> k = s*32 + (e/8)*16 + half*8 + e%8
      FragU afrag[2], bfrag[4];
#pragma unroll
      for (int mt = 0; mt < 2; ++mt) {
        int r = wmg * 32 + mt * 16 + l16;
        const bf16_t* base = &Atile[cur][r * LDS_STRIDE + s * 32 + half * 8];
        afrag[mt].u[0] = *(const uint4*)(base);
        afrag[mt].u[1] = *(const uint4*)(base + 16);
      }
#pragma unroll
      for (int nt = 0; nt < 4; ++nt) {
        int r = wng * 64 + nt * 16 + l16;
        const bf16_t* base = &Btile[cur][r * LDS_STRIDE + s * 32 + half * 8];
        bfrag[nt].u[0] = *(const uint4*)(base);
        bfrag[nt].u[1] = *(const uint4*)(base + 16);
      }
#pragma unroll
      for (int mt = 0; mt < 2; ++mt)
#pragma unroll
        for (int nt = 0; nt < 4; ++nt)
          acc[mt][nt] = __builtin_amdgcn_wmma_f32_16x16x32_bf16(
              false, afrag[mt].v, false, bfrag[nt].v,
              (short)0, acc[mt][nt], false, false);
    }

    // 3) commit next stage to the alternate buffer, single barrier per stage
    if (has_next) {
      const int nxt = cur ^ 1;
#pragma unroll
      for (int t = 0; t < 4; ++t) {
        *(uint4*)(&Atile[nxt][lofs[t]]) = va[t];
        *(uint4*)(&Btile[nxt][lofs[t]]) = vb[t];
      }
    }
    __syncthreads();
  }

  // ---- fused epilogue: sym -> sigmoid -> masks, write both outputs ----
  const float alpha = alpha_p[0];
  const float beta  = beta_p[0];
  float* logits = out;
  float* pmask  = out + (size_t)Bn * Bn;
#pragma unroll
  for (int mt = 0; mt < 2; ++mt) {
    const int mbase = m0 + wmg * 32 + mt * 16 + half * 8;  // C/D: M = r + half*8
#pragma unroll
    for (int nt = 0; nt < 4; ++nt) {
      const int n   = n0 + wng * 64 + nt * 16 + l16;       // C/D: N = lane%16
      const float sn  = srow[n];
      const float Pn  = Prow[n];
      const float rdn = rdrow[n];
      const int   ln  = labels[n];
#pragma unroll
      for (int r = 0; r < 8; ++r) {
        const int m = mbase + r;
        float g   = acc[mt][nt][r];
        float sym = 0.25f * (g + srow[m] + sn + Pn * rdrow[m] + Prow[m] * rdn
                             - 2.0f * (float)Dn);
        float x  = beta - alpha * sym;
        float lg = 1.0f / (1.0f + __expf(-x));
        bool diag = (m == n);
        size_t idx = (size_t)m * Bn + n;
        logits[idx] = diag ? 0.0f : lg;
        pmask[idx]  = (!diag && labels[m] == ln) ? 1.0f : 0.0f;
      }
    }
  }
}

extern "C" void kernel_launch(void* const* d_in, const int* in_sizes, int n_in,
                              void* d_out, int out_size, void* d_ws, size_t ws_size,
                              hipStream_t stream) {
  const float* mu      = (const float*)d_in[0];
  const float* var     = (const float*)d_in[1];
  const int*   labels  = (const int*)d_in[2];
  const float* alpha_p = (const float*)d_in[3];
  const float* beta_p  = (const float*)d_in[4];
  float* out = (float*)d_out;

  char* ws = (char*)d_ws;
  const size_t xbytes = (size_t)Bn * K4 * sizeof(bf16_t);   // 32 MB
  bf16_t* Xbf = (bf16_t*)ws;
  bf16_t* Ybf = (bf16_t*)(ws + xbytes);
  float*  s   = (float*)(ws + 2 * xbytes);
  float*  P   = (float*)(ws + 2 * xbytes + (size_t)Bn * 4);
  float*  rd  = (float*)(ws + 2 * xbytes + (size_t)Bn * 8);

  prep_kernel<<<Bn, 256, 0, stream>>>(mu, var, Xbf, Ybf, s, P, rd);
  dim3 grid(Bn / TILE, Bn / TILE);
  gemm_kernel<<<grid, 256, 0, stream>>>(Xbf, Ybf, s, P, rd, labels,
                                        alpha_p, beta_p, out);
}
